// LLaMAAttention_19086834663924
// MI455X (gfx1250) — matrix operand order repaired and checked by reading, not compiled, not run on hardware
//
#include <hip/hip_runtime.h>
#include <hip/hip_bf16.h>

#define HIDDEN   4096
#define N_HEADS  32
#define HEAD_DIM 128
#define BATCH    2
#define SEQ      2048

typedef __bf16 bf16;
typedef __attribute__((ext_vector_type(16))) __bf16       v16bf;
typedef __attribute__((ext_vector_type(8)))  float        v8f;
typedef __attribute__((ext_vector_type(4)))  unsigned int u32x4;
typedef __attribute__((ext_vector_type(2)))  unsigned int u32x2;
typedef __attribute__((ext_vector_type(4)))  int          i32x4;
typedef __attribute__((ext_vector_type(8)))  int          i32x8;

#if defined(__has_include)
#  if __has_include(<hip/amd_detail/amd_gfx1250_TDM.h>)
#    define TDM_SIX_ARG 1
#  endif
#endif

union FragBF { u32x4 u[2]; v16bf v; };
union BF4    { bf16 h[4]; u32x2 u2; };

#define WMMA_BF16(a, b, c) \
  __builtin_amdgcn_wmma_f32_16x16x32_bf16(false, (a), false, (b), (short)0, (c), false, false)

// A-matrix fragment (16x32, MxK): lane holds row M=lane&15;
// lanes 0-15 hold K {0..7,16..23}, lanes 16-31 hold K {8..15,24..31}.
__device__ __forceinline__ v16bf lds_a_frag(const bf16* t, int ld, int row0, int k0, int lane) {
  const int m = row0 + (lane & 15);
  const int klo = k0 + ((lane & 16) ? 8 : 0);
  FragBF f;
  f.u[0] = *(const u32x4*)(t + (size_t)m * ld + klo);
  f.u[1] = *(const u32x4*)(t + (size_t)m * ld + klo + 16);
  return f.v;
}

// B-matrix fragment (32x16, KxN) from a tile stored [n][k] row-major:
// lane holds column N=lane&15; lanes 0-15 K {0..7,16..23}, lanes 16-31 K {8..15,24..31}.
__device__ __forceinline__ v16bf lds_b_frag(const bf16* t, int ld, int n0, int k0, int lane) {
  const int n = n0 + (lane & 15);
  const int kk = k0 + ((lane & 16) ? 8 : 0);
  FragBF f;
  f.u[0] = *(const u32x4*)(t + (size_t)n * ld + kk);
  f.u[1] = *(const u32x4*)(t + (size_t)n * ld + kk + 16);
  return f.v;
}

// ---------------------------------------------------------------------------
// C[M,N] = A[M,K] @ W[N,K]^T  (nn.Linear).  A: f32 or bf16; C: bf16 or f32.
// 256 threads = 8 waves; block tile 128x128, K-step 64; wave tile 32x64.
// ---------------------------------------------------------------------------
template <typename AT, typename OT>
__global__ __launch_bounds__(256)
void gemm_xwt(const AT* __restrict__ A, const float* __restrict__ W,
              OT* __restrict__ C, int M, int N, int K)
{
  __shared__ bf16 sA[128 * 64];
  __shared__ bf16 sB[128 * 64];

  const int tid  = threadIdx.x;
  const int lane = tid & 31;
  const int w    = tid >> 5;
  const int wm   = w & 3;        // 4 waves along M (32 rows each)
  const int wn   = w >> 2;       // 2 waves along N (64 cols each)
  const int row0 = blockIdx.y * 128;
  const int col0 = blockIdx.x * 128;

  v8f acc[2][4] = {};

  for (int kt = 0; kt < K; kt += 64) {
    // Stage A (128x64) and W (128x64) tiles into LDS as bf16, packed 4-wide.
    #pragma unroll
    for (int i = 0; i < 8; ++i) {
      const int lin = (tid + i * 256) * 4;
      const int r = lin >> 6, c = lin & 63;
      if constexpr (sizeof(AT) == 4) {
        const float4 v = *(const float4*)((const float*)A + (size_t)(row0 + r) * K + kt + c);
        BF4 t;
        t.h[0] = (bf16)v.x; t.h[1] = (bf16)v.y; t.h[2] = (bf16)v.z; t.h[3] = (bf16)v.w;
        *(u32x2*)(sA + r * 64 + c) = t.u2;
      } else {
        *(u32x2*)(sA + r * 64 + c) =
            *(const u32x2*)((const bf16*)A + (size_t)(row0 + r) * K + kt + c);
      }
      const float4 wv = *(const float4*)(W + (size_t)(col0 + r) * K + kt + c);
      BF4 t;
      t.h[0] = (bf16)wv.x; t.h[1] = (bf16)wv.y; t.h[2] = (bf16)wv.z; t.h[3] = (bf16)wv.w;
      *(u32x2*)(sB + r * 64 + c) = t.u2;
    }
    if (kt + 64 < K) {  // hint next K-tile of the weight into cache
      __builtin_prefetch(W + (size_t)(col0 + (tid >> 1)) * K + kt + 64 + (tid & 1) * 32, 0, 1);
    }
    __syncthreads();

    #pragma unroll
    for (int ks = 0; ks < 2; ++ks) {
      v16bf af[2], bfm[4];
      #pragma unroll
      for (int m = 0; m < 2; ++m) af[m] = lds_a_frag(sA, 64, wm * 32 + m * 16, ks * 32, lane);
      #pragma unroll
      for (int n = 0; n < 4; ++n) bfm[n] = lds_b_frag(sB, 64, wn * 64 + n * 16, ks * 32, lane);
      #pragma unroll
      for (int m = 0; m < 2; ++m)
        #pragma unroll
        for (int n = 0; n < 4; ++n)
          acc[m][n] = WMMA_BF16(af[m], bfm[n], acc[m][n]);
    }
    __syncthreads();
  }

  // C/D layout: VGPR j -> row j (lanes 0-15) / row j+8 (lanes 16-31); col = lane&15.
  const int rbase = row0 + wm * 32 + ((lane & 16) ? 8 : 0);
  const int cbase = col0 + wn * 64 + (lane & 15);
  #pragma unroll
  for (int m = 0; m < 2; ++m)
    #pragma unroll
    for (int n = 0; n < 4; ++n)
      #pragma unroll
      for (int j = 0; j < 8; ++j)
        C[(size_t)(rbase + m * 16 + j) * N + cbase + n * 16] = (OT)acc[m][n][j];
}

// ---------------------------------------------------------------------------
// In-place RoPE over Q or K (blockIdx.z selects). One thread per (d, d+64) pair.
// ---------------------------------------------------------------------------
__global__ __launch_bounds__(256)
void rope_inplace(bf16* __restrict__ q, bf16* __restrict__ k)
{
  bf16* buf = blockIdx.z ? k : q;
  const int gid = blockIdx.x * 256 + threadIdx.x;
  const int p  = gid & 63;              // pair index within head
  const int hh = (gid >> 6) & (N_HEADS - 1);
  const int bs = gid >> 11;             // flattened (b, s)
  const int s  = bs & (SEQ - 1);
  const size_t base = (size_t)bs * HIDDEN + hh * HEAD_DIM;
  const float inv = __powf(10000.0f, -(float)p * (1.0f / 64.0f));
  float sn, cs;
  __sincosf((float)s * inv, &sn, &cs);
  const float x0 = (float)buf[base + p];
  const float x1 = (float)buf[base + p + 64];
  buf[base + p]      = (bf16)(x0 * cs - x1 * sn);
  buf[base + p + 64] = (bf16)(x1 * cs + x0 * sn);
}

// ---------------------------------------------------------------------------
// Streaming-softmax attention. Block = 8 waves, each wave owns 16 q rows.
// K tile staged by the Tensor Data Mover (tensor_load_to_lds); V transposed
// manually. Per kv-tile: S = Q K^T (16 WMMA), online softmax, O += P V (16).
// ---------------------------------------------------------------------------
__global__ __launch_bounds__(256)
void flash_attn(const bf16* __restrict__ Q, const bf16* __restrict__ K,
                const bf16* __restrict__ V, bf16* __restrict__ O)
{
  __shared__ bf16 sK[64 * 128];     // K tile, [kv][d]
  __shared__ bf16 sVt[128 * 72];    // V tile transposed, [d][kv] (+pad)
  __shared__ bf16 sP[8][16 * 64];   // per-wave P relayout buffer

  const int tid  = threadIdx.x;
  const int lane = tid & 31;
  const int w    = tid >> 5;
  const int h    = blockIdx.y;
  const int b    = blockIdx.z;
  const int q0   = blockIdx.x * 128 + w * 16;

  const size_t headoff = (size_t)(b * SEQ) * HIDDEN + (size_t)h * HEAD_DIM;
  const bf16* qb = Q + headoff;
  const bf16* kb = K + headoff;
  const bf16* vb = V + headoff;

  // Q fragments: 16 rows x 128 d, four K-steps of 32 (A layout).
  v16bf qf[4];
  {
    const bf16* qp = qb + (size_t)(q0 + (lane & 15)) * HIDDEN;
    const int klo = (lane & 16) ? 8 : 0;
    #pragma unroll
    for (int f = 0; f < 4; ++f) {
      FragBF t;
      t.u[0] = *(const u32x4*)(qp + f * 32 + klo);
      t.u[1] = *(const u32x4*)(qp + f * 32 + klo + 16);
      qf[f] = t.v;
    }
  }

  const float SCL = 0.12751369543f;  // log2(e) / sqrt(HEAD_DIM)
  float mrow[8], lrow[8];
  #pragma unroll
  for (int j = 0; j < 8; ++j) { mrow[j] = -1e30f; lrow[j] = 0.0f; }
  v8f o[8] = {};

  for (int kv = 0; kv < SEQ; kv += 64) {
    __syncthreads();  // protect sK/sVt reuse across iterations

#if defined(__gfx1250__) && __has_builtin(__builtin_amdgcn_tensor_load_to_lds)
    // --- TDM: DMA the 64x128 bf16 K tile (row stride HIDDEN) into sK. ---
    if (w == 0) {
      const unsigned lds_off = (unsigned)(size_t)(void*)sK;  // low 32b = LDS offset
      const unsigned long long ga =
          (unsigned long long)(size_t)(const void*)(kb + (size_t)kv * HIDDEN);
      u32x4 g0;
      g0.x = 1u;                                              // count=1: valid user D#
      g0.y = lds_off;                                         // lds_addr
      g0.z = (unsigned)ga;                                    // global_addr[31:0]
      g0.w = (unsigned)((ga >> 32) & 0x01FFFFFFu) | (2u << 30); // addr[56:32] | type=2
      i32x8 g1;
      g1[0] = (int)(1u << 16);                                // data_size = 2 bytes
      g1[1] = (int)(((unsigned)HEAD_DIM & 0xFFFFu) << 16);    // tensor_dim0 lo -> [63:48]
      g1[2] = (int)((64u & 0xFFFFu) << 16);                   // tdim0 hi=0 | tensor_dim1 lo
      g1[3] = (int)((unsigned)HEAD_DIM << 16);                // tdim1 hi=0 | tile_dim0=128
      g1[4] = (int)64u;                                       // tile_dim1=64, tile_dim2=0
      g1[5] = (int)HIDDEN;                                    // tensor_dim0_stride lo
      g1[6] = 0;                                              // stride0 hi | stride1 lo
      g1[7] = 0;                                              // stride1 hi
      const i32x4 gz = {0, 0, 0, 0};
#ifdef TDM_SIX_ARG
      const i32x8 gz8 = {0, 0, 0, 0, 0, 0, 0, 0};
      __builtin_amdgcn_tensor_load_to_lds(g0, g1, gz, gz, gz8, 0);
#else
      __builtin_amdgcn_tensor_load_to_lds(g0, g1, gz, gz, 0);
#endif
      __builtin_amdgcn_s_wait_tensorcnt(0);
    }
#else
    // Fallback: manual K staging.
    #pragma unroll
    for (int i = 0; i < 4; ++i) {
      const int lin = (tid + i * 256) * 8;
      const int r = lin >> 7, c = lin & 127;
      *(u32x4*)(sK + r * 128 + c) = *(const u32x4*)(kb + (size_t)(kv + r) * HIDDEN + c);
    }
#endif

    // V staged transposed (coalesced global read, scattered LDS write).
    #pragma unroll
    for (int i = 0; i < 4; ++i) {
      const int lin = (tid + i * 256) * 8;
      const int r = lin >> 7, c = lin & 127;      // r = kv row, c = d
      const bf16* vp = vb + (size_t)(kv + r) * HIDDEN + c;
      #pragma unroll
      for (int e = 0; e < 8; ++e) sVt[(c + e) * 72 + r] = vp[e];
    }
    if (kv + 64 < SEQ) {
      __builtin_prefetch(vb + (size_t)(kv + 64 + (tid >> 2)) * HIDDEN + (tid & 3) * 32, 0, 1);
    }
    __syncthreads();

    // S = Q K^T : 16 rows x 64 kv
    v8f s[4] = {};
    #pragma unroll
    for (int n = 0; n < 4; ++n)
      #pragma unroll
      for (int ks = 0; ks < 4; ++ks)
        s[n] = WMMA_BF16(qf[ks], lds_b_frag(sK, 128, n * 16, ks * 32, lane), s[n]);

    // Row max (reduce across the 16-lane half holding each row's 16 columns).
    float mx[8];
    #pragma unroll
    for (int j = 0; j < 8; ++j) mx[j] = -1e30f;
    #pragma unroll
    for (int n = 0; n < 4; ++n)
      #pragma unroll
      for (int j = 0; j < 8; ++j) mx[j] = fmaxf(mx[j], s[n][j]);
    #pragma unroll
    for (int j = 0; j < 8; ++j)
      #pragma unroll
      for (int d = 1; d < 16; d <<= 1)
        mx[j] = fmaxf(mx[j], __shfl_xor(mx[j], d, 32));

    float rsc[8], sum[8];
    #pragma unroll
    for (int j = 0; j < 8; ++j) {
      const float mn = fmaxf(mrow[j], mx[j]);
      rsc[j] = exp2f((mrow[j] - mn) * SCL);
      mrow[j] = mn;
      sum[j] = 0.0f;
    }

    // P = exp2((S - m) * scale); spill bf16 P through LDS to get A layout.
    const int prow = (lane & 16) ? 8 : 0;
    #pragma unroll
    for (int n = 0; n < 4; ++n)
      #pragma unroll
      for (int j = 0; j < 8; ++j) {
        const float p = exp2f((s[n][j] - mrow[j]) * SCL);
        sum[j] += p;
        sP[w][(prow + j) * 64 + n * 16 + (lane & 15)] = (bf16)p;
      }
    #pragma unroll
    for (int j = 0; j < 8; ++j) {
      #pragma unroll
      for (int d = 1; d < 16; d <<= 1)
        sum[j] += __shfl_xor(sum[j], d, 32);
      lrow[j] = lrow[j] * rsc[j] + sum[j];
    }
    #pragma unroll
    for (int dd = 0; dd < 8; ++dd)
      #pragma unroll
      for (int j = 0; j < 8; ++j) o[dd][j] *= rsc[j];

    // Reload P as A fragments (wave-local LDS; DS ops are in-order per wave).
    v16bf pf[2];
    {
      const int m = lane & 15;
      const int klo = (lane & 16) ? 8 : 0;
      #pragma unroll
      for (int ks = 0; ks < 2; ++ks) {
        FragBF t;
        t.u[0] = *(const u32x4*)(&sP[w][m * 64 + ks * 32 + klo]);
        t.u[1] = *(const u32x4*)(&sP[w][m * 64 + ks * 32 + klo + 16]);
        pf[ks] = t.v;
      }
    }
    // O += P @ V
    #pragma unroll
    for (int dd = 0; dd < 8; ++dd)
      #pragma unroll
      for (int ks = 0; ks < 2; ++ks)
        o[dd] = WMMA_BF16(pf[ks], lds_b_frag(sVt, 72, dd * 16, ks * 32, lane), o[dd]);
  }

  // Normalize and store bf16 attention output in (b, s, h*128+d) layout.
  bf16* ob = O + headoff;
  const int rbase = q0 + ((lane & 16) ? 8 : 0);
  float invl[8];
  #pragma unroll
  for (int j = 0; j < 8; ++j) invl[j] = 1.0f / lrow[j];
  #pragma unroll
  for (int dd = 0; dd < 8; ++dd)
    #pragma unroll
    for (int j = 0; j < 8; ++j)
      ob[(size_t)(rbase + j) * HIDDEN + dd * 16 + (lane & 15)] = (bf16)(o[dd][j] * invl[j]);
}

// ---------------------------------------------------------------------------
extern "C" void kernel_launch(void* const* d_in, const int* in_sizes, int n_in,
                              void* d_out, int out_size, void* d_ws, size_t ws_size,
                              hipStream_t stream) {
  (void)in_sizes; (void)n_in; (void)out_size; (void)ws_size;
  const float* X  = (const float*)d_in[0];
  const float* Wq = (const float*)d_in[1];
  const float* Wk = (const float*)d_in[2];
  const float* Wv = (const float*)d_in[3];
  const float* Wo = (const float*)d_in[4];

  const size_t elems = (size_t)BATCH * SEQ * HIDDEN;
  bf16* qbuf = (bf16*)d_ws;          // 32 MB each
  bf16* kbuf = qbuf + elems;
  bf16* vbuf = kbuf + elems;
  bf16* obuf = vbuf + elems;

  const dim3 blk(256);
  const dim3 gg(HIDDEN / 128, (BATCH * SEQ) / 128);

  gemm_xwt<float, bf16><<<gg, blk, 0, stream>>>(X, Wq, qbuf, BATCH * SEQ, HIDDEN, HIDDEN);
  gemm_xwt<float, bf16><<<gg, blk, 0, stream>>>(X, Wk, kbuf, BATCH * SEQ, HIDDEN, HIDDEN);
  gemm_xwt<float, bf16><<<gg, blk, 0, stream>>>(X, Wv, vbuf, BATCH * SEQ, HIDDEN, HIDDEN);

  rope_inplace<<<dim3((BATCH * SEQ * N_HEADS * 64) / 256, 1, 2), blk, 0, stream>>>(qbuf, kbuf);

  flash_attn<<<dim3(SEQ / 128, N_HEADS, BATCH), blk, 0, stream>>>(qbuf, kbuf, vbuf, obuf);

  gemm_xwt<bf16, float><<<gg, blk, 0, stream>>>(obuf, Wo, (float*)d_out,
                                                BATCH * SEQ, HIDDEN, HIDDEN);
}
